// SwinTransformerBlock_4337916970084
// MI455X (gfx1250) — matrix operand order repaired and checked by reading, not compiled, not run on hardware
//
#include <hip/hip_runtime.h>
#include <hip/hip_bf16.h>

typedef _Float16 half_t;
typedef __attribute__((ext_vector_type(16))) _Float16 v16h;
typedef __attribute__((ext_vector_type(8)))  _Float16 v8h;
typedef __attribute__((ext_vector_type(8)))  float    v8f;

#define DEV static __device__ __forceinline__

constexpr int DIMC  = 180;
constexpr int LD    = 192;                 // padded feature stride
constexpr int NH    = 6;
constexpr int HD    = 30;
constexpr int NTOK  = 4 * 256 * 256;       // 262144 tokens
constexpr float LN100 = 4.6051701859880914f;

// ---------------------------------------------------------------------------
// WMMA fragment helpers (v_wmma_f32_16x16x32_f16, wave32).
// All buffers are zero-padded so fragment loads are unconditional wide loads.
// ---------------------------------------------------------------------------
DEV v8f wmma_f16(v16h a, v16h b, v8f c) {
  return __builtin_amdgcn_wmma_f32_16x16x32_f16(false, a, false, b, (short)0, c, false, false);
}

DEV v16h join8(v8h lo, v8h hi) {
  v16h f;
#pragma unroll
  for (int i = 0; i < 8; ++i) { f[i] = lo[i]; f[8 + i] = hi[i]; }
  return f;
}

// A fragment from row-major [rows x ld]; ld multiple of 8, k0 multiple of 32.
DEV v16h frag_A(const half_t* base, int ld, int row0, int k0) {
  const int lane = threadIdx.x & 31;
  const half_t* rp = base + (size_t)(row0 + (lane & 15)) * ld + k0 + ((lane >> 4) << 3);
  return join8(*(const v8h*)rp, *(const v8h*)(rp + 16));
}

// A fragment for implicit-GEMM conv: token rows shifted by (dh,dw); out-of-image
// rows read a zero row (pointer select, no per-element predication).
DEV v16h frag_A_conv(const half_t* base, const half_t* zrow, int row0,
                     int dh, int dw, int k0, int ld) {
  const int lane = threadIdx.x & 31;
  const int tok = row0 + (lane & 15);
  const int b = tok >> 16, hw = tok & 65535;
  const int h = (hw >> 8) + dh, w = (hw & 255) + dw;
  const bool valid = ((unsigned)h < 256u) && ((unsigned)w < 256u);
  const half_t* rp = valid ? (base + (((size_t)b << 16) + (size_t)(h * 256 + w)) * ld) : zrow;
  rp += k0 + ((lane >> 4) << 3);
  return join8(*(const v8h*)rp, *(const v8h*)(rp + 16));
}

// B fragment = W^T, W row-major [N x ld]; two 16B loads per lane (same k order as frag_A).
DEV v16h frag_BT(const half_t* W, int ld, int n0, int k0) {
  const int lane = threadIdx.x & 31;
  const half_t* rp = W + (size_t)(n0 + (lane & 15)) * ld + k0 + ((lane >> 4) << 3);
  return join8(*(const v8h*)rp, *(const v8h*)(rp + 16));
}

// ---------------------------------------------------------------------------
// weight / bias prep kernels (f32 -> f16, padded layouts)
// ---------------------------------------------------------------------------
__global__ void k_wpad(const float* __restrict__ src, half_t* __restrict__ dst,
                       int R, int K, int Rout, int Kout) {
  int i = blockIdx.x * 256 + threadIdx.x;
  if (i >= Rout * Kout) return;
  int r = i / Kout, k = i % Kout;
  dst[i] = (r < R && k < K) ? (half_t)src[r * K + k] : (half_t)0;
}

// qkv_w (540,180) -> [3][6][32][192], rows 30..31 and cols 180..191 zero
__global__ void k_wqkv(const float* __restrict__ src, half_t* __restrict__ dst) {
  int i = blockIdx.x * 256 + threadIdx.x;
  if (i >= 3 * 6 * 32 * LD) return;
  int k = i % LD, r = (i / LD) & 31, mh = i / (LD * 32);
  int mat = mh / 6, head = mh % 6;
  float v = 0.f;
  if (r < HD && k < DIMC) v = src[(mat * DIMC + head * HD + r) * DIMC + k];
  dst[i] = (half_t)v;
}

// cab_c1_w (60,180,3,3) -> [9][64][192]
__global__ void k_w1r(const float* __restrict__ src, half_t* __restrict__ dst) {
  int i = blockIdx.x * 256 + threadIdx.x;
  if (i >= 9 * 64 * LD) return;
  int ic = i % LD, oc = (i / LD) & 63, tap = i / (LD * 64);
  float v = 0.f;
  if (oc < 60 && ic < DIMC) v = src[(oc * DIMC + ic) * 9 + tap];
  dst[i] = (half_t)v;
}

// cab_c2_w (180,60,3,3) -> [9][192][64]
__global__ void k_w2r(const float* __restrict__ src, half_t* __restrict__ dst) {
  int i = blockIdx.x * 256 + threadIdx.x;
  if (i >= 9 * LD * 64) return;
  int ic = i % 64, oc = (i / 64) % LD, tap = i / (64 * LD);
  float v = 0.f;
  if (oc < DIMC && ic < 60) v = src[(oc * 60 + ic) * 9 + tap];
  dst[i] = (half_t)v;
}

// padded bias vectors: qkvb[3][6][32], cb1p[64], cb2p[192], pbp[192], f2bp[192]
__global__ void k_biases(const float* __restrict__ qb, const float* __restrict__ vb,
                         const float* __restrict__ c1b, const float* __restrict__ c2b,
                         const float* __restrict__ pb, const float* __restrict__ f2b,
                         float* __restrict__ qkvb, float* __restrict__ cb1p,
                         float* __restrict__ cb2p, float* __restrict__ pbp,
                         float* __restrict__ f2bp) {
  int i = blockIdx.x * 256 + threadIdx.x;
  if (i < 3 * NH * 32) {
    int mat = i / (NH * 32), head = (i >> 5) % NH, r = i & 31;
    float v = 0.f;
    if (r < HD) {
      if (mat == 0) v = qb[head * HD + r];
      else if (mat == 2) v = vb[head * HD + r];
    }
    qkvb[i] = v;
  }
  if (i < 64)  cb1p[i] = (i < 60)   ? c1b[i] : 0.f;
  if (i < LD)  cb2p[i] = (i < DIMC) ? c2b[i] : 0.f;
  if (i < LD)  pbp[i]  = (i < DIMC) ? pb[i]  : 0.f;
  if (i < LD)  f2bp[i] = (i < DIMC) ? f2b[i] : 0.f;
}

__global__ void k_zero_h(half_t* __restrict__ p, int n) {
  int i = blockIdx.x * 256 + threadIdx.x;
  if (i < n) p[i] = (half_t)0;
}

// zero pad columns [180..191] of a [NTOK x 192] f16 buffer
__global__ void k_padcols(half_t* __restrict__ p) {
  int i = blockIdx.x * 256 + threadIdx.x;
  if (i >= NTOK * 12) return;
  p[(size_t)(i / 12) * LD + DIMC + (i % 12)] = (half_t)0;
}

DEV float table_val(int a) {
  float v = (float)(a - 7) * (8.0f / 7.0f);
  float s = (v > 0.f) ? 1.f : ((v < 0.f) ? -1.f : 0.f);
  return s * log2f(fabsf(v) + 1.f) * (1.0f / log2f(8.0f));
}

// CPB MLP -> bias16[h][n][m] = 16*sigmoid(rpb)
__global__ void k_cpb(const float* __restrict__ w1, const float* __restrict__ b1,
                      const float* __restrict__ w2, float* __restrict__ bias16) {
  __shared__ float t[225 * 6];
  for (int i = threadIdx.x; i < 225 * 6; i += blockDim.x) {
    int idx = i / 6, h = i % 6;
    float ca = table_val(idx / 15), cb = table_val(idx % 15);
    float acc = 0.f;
    for (int j = 0; j < 512; ++j) {
      float hv = fmaxf(ca * w1[j * 2 + 0] + cb * w1[j * 2 + 1] + b1[j], 0.f);
      acc += hv * w2[h * 512 + j];
    }
    t[i] = acc;
  }
  __syncthreads();
  for (int i = threadIdx.x; i < 6 * 64 * 64; i += blockDim.x) {
    int h = i / 4096, rem = i % 4096;
    int n = rem / 64, m = rem % 64;
    int idx = ((n >> 3) - (m >> 3) + 7) * 15 + ((n & 7) - (m & 7) + 7);
    bias16[i] = 16.f / (1.f + __expf(-t[idx * 6 + h]));
  }
}

// LayerNorm (wave-per-token), f32 in -> f16 out with zero pad cols
__global__ __launch_bounds__(256) void k_ln(const float* __restrict__ x,
                                            const float* __restrict__ g,
                                            const float* __restrict__ bb,
                                            half_t* __restrict__ out) {
  int wid = blockIdx.x * (blockDim.x >> 5) + (threadIdx.x >> 5);
  if (wid >= NTOK) return;
  int lane = threadIdx.x & 31;
  const float* row = x + (size_t)wid * DIMC;
  float s = 0.f, s2 = 0.f;
  for (int j = lane; j < DIMC; j += 32) { float v = row[j]; s += v; s2 += v * v; }
  for (int off = 16; off > 0; off >>= 1) { s += __shfl_xor(s, off, 32); s2 += __shfl_xor(s2, off, 32); }
  float mean = s / (float)DIMC;
  float var  = s2 / (float)DIMC - mean * mean;
  float inv  = rsqrtf(var + 1e-5f);
  half_t* orow = out + (size_t)wid * LD;
  for (int j = lane; j < LD; j += 32) {
    float v = (j < DIMC) ? ((row[j] - mean) * inv * g[j] + bb[j]) : 0.f;
    orow[j] = (half_t)v;
  }
}

DEV int label_of(int wh, int ww, int t) {
  int hh = wh * 8 + (t >> 3);
  int wc = ww * 8 + (t & 7);
  int rl = (hh < 248) ? 0 : ((hh < 252) ? 1 : 2);
  int cl = (wc < 248) ? 0 : ((wc < 252) ? 1 : 2);
  return rl * 3 + cl;
}

// one statically-specialized qkv pass: MAT=0 -> sq, 1 -> sk, 2 -> svT (transposed)
template <int MAT>
DEV void qkv_pass(const half_t* sx, const half_t* __restrict__ wqkv,
                  const float* __restrict__ qkvb, int head, int wv,
                  half_t* dstQK, half_t (*svT)[64]) {
  const int rt = wv >> 1, ct = wv & 1;                 // 8 waves = 4x2 tiles
  const half_t* wmat = wqkv + (size_t)((MAT * NH + head) * 32) * LD;
  v8f acc = {};
#pragma unroll
  for (int ks = 0; ks < 6; ++ks) {
    v16h a  = frag_A(sx, LD, rt * 16, ks * 32);
    v16h bf = frag_BT(wmat, LD, ct * 16, ks * 32);
    acc = wmma_f16(a, bf, acc);
  }
  const int lane = threadIdx.x & 31;
  const int n  = ct * 16 + (lane & 15);
  const int mb = rt * 16 + ((lane >> 4) << 3);
  // padded bias: zero weight rows + zero bias pad guarantee 0 in pad lanes
  const float bias = qkvb[(MAT * NH + head) * 32 + n];
  if (MAT == 2) {                                      // compile-time branch
    v8h pk;
#pragma unroll
    for (int v = 0; v < 8; ++v) pk[v] = (half_t)(acc[v] + bias);
    *(v8h*)&svT[n][mb] = pk;                           // one 16B ds_store
  } else {
#pragma unroll
    for (int v = 0; v < 8; ++v)
      dstQK[(mb + v) * 32 + n] = (half_t)(acc[v] + bias);
  }
}

// ---------------------------------------------------------------------------
// fused window attention: one block per (window, head)
// ---------------------------------------------------------------------------
__global__ __launch_bounds__(256) void k_attn(const half_t* __restrict__ xn,
                                              const half_t* __restrict__ wqkv,
                                              const float* __restrict__ qkvb,
                                              const float* __restrict__ lsc_in,
                                              const float* __restrict__ bias16,
                                              half_t* __restrict__ attn_out) {
  __shared__ alignas(32) half_t sx[64][LD];
  __shared__ alignas(32) half_t sq[64][32], sk[64][32];
  __shared__ alignas(32) half_t svT[32][64];          // v transposed: [dim][token]
  __shared__ float  sattn[64][64];
  __shared__ alignas(32) half_t sattn_h[64][64];

  const int head = blockIdx.x % NH;
  const int w    = blockIdx.x / NH;
  const int b    = w >> 10;
  const int widx = w & 1023;
  const int wh   = widx >> 5, ww = widx & 31;
  const int wv   = threadIdx.x >> 5;

  // stage shifted window tokens, 16B vector copies (roll -4 folded into gather)
  for (int i = threadIdx.x; i < 64 * (LD / 8); i += 256) {   // 64*24 uint4s
    int t = i / 24, j = i % 24;
    int h0 = (wh * 8 + (t >> 3) + 4) & 255;
    int w0 = (ww * 8 + (t & 7) + 4) & 255;
    size_t tok = ((size_t)b << 16) + (size_t)(h0 * 256 + w0);
    ((uint4*)&sx[0][0])[t * 24 + j] = ((const uint4*)xn)[tok * 24 + j];
  }
  __syncthreads();

  // qkv GEMM for this head: three uniform passes (no data-dependent stores)
  qkv_pass<0>(&sx[0][0], wqkv, qkvb, head, wv, &sq[0][0], svT);
  qkv_pass<1>(&sx[0][0], wqkv, qkvb, head, wv, &sk[0][0], svT);
  qkv_pass<2>(&sx[0][0], wqkv, qkvb, head, wv, nullptr,   svT);
  __syncthreads();

  // cosine normalize rows of q,k ; fold clamped logit scale into q
  const float lsc = __expf(fminf(lsc_in[head], LN100));
  for (int i = threadIdx.x; i < 128; i += 256) {
    int row = i & 63;
    half_t* mat = (i < 64) ? &sq[0][0] : &sk[0][0];
    float ss = 0.f;
    for (int j = 0; j < HD; ++j) { float v = (float)mat[row * 32 + j]; ss += v * v; }
    float sc = 1.0f / fmaxf(sqrtf(ss), 1e-12f);
    if (i < 64) sc *= lsc;
    for (int j = 0; j < HD; ++j) mat[row * 32 + j] = (half_t)((float)mat[row * 32 + j] * sc);
  }
  __syncthreads();

  // attn = q @ k^T  (4x4 tiles, K=32) ; + CPB bias + shift mask
  for (int tile = wv; tile < 16; tile += 8) {
    const int rt = tile >> 2, ct = tile & 3;
    v8f acc = {};
    v16h a  = frag_A(&sq[0][0], 32, rt * 16, 0);
    v16h bf = frag_BT(&sk[0][0], 32, ct * 16, 0);
    acc = wmma_f16(a, bf, acc);
    const int lane = threadIdx.x & 31;
    const int kj = ct * 16 + (lane & 15);
    const int mb = rt * 16 + ((lane >> 4) << 3);
    const int lk = label_of(wh, ww, kj);
#pragma unroll
    for (int v = 0; v < 8; ++v) {
      const int qi = mb + v;
      float bb = bias16[((head << 6) + qi) * 64 + kj];
      float mk = (label_of(wh, ww, qi) == lk) ? 0.f : -100.f;
      sattn[qi][kj] = acc[v] + bb + mk;
    }
  }
  __syncthreads();

  // row softmax
  for (int row = threadIdx.x; row < 64; row += 256) {
    float mx = -1e30f;
    for (int j = 0; j < 64; ++j) mx = fmaxf(mx, sattn[row][j]);
    float s = 0.f;
    for (int j = 0; j < 64; ++j) { float e = __expf(sattn[row][j] - mx); sattn[row][j] = e; s += e; }
    float inv = 1.f / s;
    for (int j = 0; j < 64; ++j) sattn_h[row][j] = (half_t)(sattn[row][j] * inv);
  }
  __syncthreads();

  // out = attn @ v  (4x2 tiles, K=64) -> scatter with reverse-roll mapping
  {
    const int tile = wv;               // exactly 8 tiles
    const int rt = tile >> 1, ct = tile & 1;
    v8f acc = {};
#pragma unroll
    for (int ks = 0; ks < 2; ++ks) {
      v16h a  = frag_A(&sattn_h[0][0], 64, rt * 16, ks * 32);
      v16h bf = frag_BT(&svT[0][0], 64, ct * 16, ks * 32);   // B[k][n] = svT[n][k]
      acc = wmma_f16(a, bf, acc);
    }
    const int lane = threadIdx.x & 31;
    const int n  = ct * 16 + (lane & 15);
    const int mb = rt * 16 + ((lane >> 4) << 3);
    if (n < HD) {
#pragma unroll
      for (int v = 0; v < 8; ++v) {
        const int t = mb + v;
        int h0 = (wh * 8 + (t >> 3) + 4) & 255;
        int w0 = (ww * 8 + (t & 7) + 4) & 255;
        size_t tok = ((size_t)b << 16) + (size_t)(h0 * 256 + w0);
        attn_out[tok * LD + head * HD + n] = (half_t)acc[v];
      }
    }
  }
}

// ---------------------------------------------------------------------------
// implicit-GEMM 3x3 convs (WMMA over 9 taps)
// ---------------------------------------------------------------------------
__global__ __launch_bounds__(256) void k_conv1(const half_t* __restrict__ xn,
                                               const half_t* __restrict__ w1,
                                               const half_t* __restrict__ zrow,
                                               const float* __restrict__ b1p,
                                               half_t* __restrict__ c1) {
  const int row0 = blockIdx.x * 64;
  const int wv = threadIdx.x >> 5;
  for (int tile = wv; tile < 16; tile += 8) {          // 4 row x 4 col tiles (64 oc)
    const int rt = tile >> 2, ct = tile & 3;
    v8f acc = {};
    for (int tap = 0; tap < 9; ++tap) {
      const int dh = tap / 3 - 1, dw = tap % 3 - 1;
#pragma unroll
      for (int ks = 0; ks < 6; ++ks) {
        v16h a  = frag_A_conv(xn, zrow, row0 + rt * 16, dh, dw, ks * 32, LD);
        v16h bf = frag_BT(w1 + (size_t)tap * 64 * LD, LD, ct * 16, ks * 32);
        acc = wmma_f16(a, bf, acc);
      }
    }
    const int lane = threadIdx.x & 31;
    const int oc = ct * 16 + (lane & 15);
    const int mb = row0 + rt * 16 + ((lane >> 4) << 3);
    const float bias = b1p[oc];
#pragma unroll
    for (int v = 0; v < 8; ++v) {
      float xv = acc[v] + bias;                       // pad lanes: 0
      float g = 0.5f * xv * (1.f + erff(xv * 0.70710678118654752f));   // gelu(0)=0
      c1[(size_t)(mb + v) * 64 + oc] = (half_t)g;
    }
  }
}

__global__ __launch_bounds__(256) void k_conv2(const half_t* __restrict__ c1,
                                               const half_t* __restrict__ w2,
                                               const half_t* __restrict__ zrow,
                                               const float* __restrict__ b2p,
                                               half_t* __restrict__ cfeat) {
  const int row0 = blockIdx.x * 64;
  const int wv = threadIdx.x >> 5;
  for (int tile = wv; tile < 48; tile += 8) {          // 4 row x 12 col tiles (192 oc)
    const int rt = tile / 12, ct = tile % 12;
    v8f acc = {};
    for (int tap = 0; tap < 9; ++tap) {
      const int dh = tap / 3 - 1, dw = tap % 3 - 1;
#pragma unroll
      for (int ks = 0; ks < 2; ++ks) {
        v16h a  = frag_A_conv(c1, zrow, row0 + rt * 16, dh, dw, ks * 32, 64);
        v16h bf = frag_BT(w2 + (size_t)tap * LD * 64, 64, ct * 16, ks * 32);
        acc = wmma_f16(a, bf, acc);
      }
    }
    const int lane = threadIdx.x & 31;
    const int oc = ct * 16 + (lane & 15);
    const int mb = row0 + rt * 16 + ((lane >> 4) << 3);
    const float bias = b2p[oc];
#pragma unroll
    for (int v = 0; v < 8; ++v)                        // pad cols store 0
      cfeat[(size_t)(mb + v) * LD + oc] = (half_t)(acc[v] + bias);
  }
}

__global__ __launch_bounds__(256) void k_cmean(const half_t* __restrict__ c, float* __restrict__ s) {
  __shared__ float red[256];
  const int b = blockIdx.x / DIMC, ch = blockIdx.x % DIMC;
  float acc = 0.f;
  for (int i = threadIdx.x; i < 65536; i += 256)
    acc += (float)c[(((size_t)b << 16) + i) * LD + ch];
  red[threadIdx.x] = acc; __syncthreads();
  for (int st = 128; st > 0; st >>= 1) {
    if ((int)threadIdx.x < st) red[threadIdx.x] += red[threadIdx.x + st];
    __syncthreads();
  }
  if (threadIdx.x == 0) s[blockIdx.x] = red[0] * (1.0f / 65536.0f);
}

__global__ void k_ca(const float* __restrict__ s, const float* __restrict__ w1,
                     const float* __restrict__ b1, const float* __restrict__ w2,
                     const float* __restrict__ b2, float* __restrict__ a) {
  __shared__ float hid[4][6];
  if (threadIdx.x < 24) {
    int b = threadIdx.x / 6, o = threadIdx.x % 6;
    float acc = b1[o];
    for (int ch = 0; ch < DIMC; ++ch) acc += s[b * DIMC + ch] * w1[o * DIMC + ch];
    hid[b][o] = fmaxf(acc, 0.f);
  }
  __syncthreads();
  for (int i = threadIdx.x; i < 4 * DIMC; i += blockDim.x) {
    int b = i / DIMC, ch = i % DIMC;
    float acc = b2[ch];
    for (int o = 0; o < 6; ++o) acc += hid[b][o] * w2[ch * 6 + o];
    a[i] = 1.f / (1.f + __expf(-acc));
  }
}

// ---------------------------------------------------------------------------
// proj GEMM + fused residual: out = x + attn@Wp^T + pb + 0.01*c*a
// ---------------------------------------------------------------------------
__global__ __launch_bounds__(256) void k_proj(const half_t* __restrict__ attn_in,
                                              const half_t* __restrict__ wproj,
                                              const float* __restrict__ pbp,
                                              const float* __restrict__ x,
                                              const half_t* __restrict__ cfeat,
                                              const float* __restrict__ aca,
                                              float* __restrict__ out) {
  const int row0 = blockIdx.x * 64;
  const int wv = threadIdx.x >> 5;
  for (int tile = wv; tile < 48; tile += 8) {
    const int rt = tile / 12, ct = tile % 12;
    v8f acc = {};
#pragma unroll
    for (int ks = 0; ks < 6; ++ks) {
      v16h a  = frag_A(attn_in, LD, row0 + rt * 16, ks * 32);
      v16h bf = frag_BT(wproj, LD, ct * 16, ks * 32);
      acc = wmma_f16(a, bf, acc);
    }
    const int lane = threadIdx.x & 31;
    const int oc = ct * 16 + (lane & 15);
    const int mb = row0 + rt * 16 + ((lane >> 4) << 3);
    const float bias = pbp[oc];
    if (oc < DIMC) {                                   // d_out is packed ld=180
#pragma unroll
      for (int v = 0; v < 8; ++v) {
        size_t tok = (size_t)(mb + v);
        int b = (int)(tok >> 16);
        float cv = (float)cfeat[tok * LD + oc];
        out[tok * DIMC + oc] = x[tok * DIMC + oc] + acc[v] + bias
                             + 0.01f * cv * aca[b * DIMC + oc];
      }
    }
  }
}

// ---------------------------------------------------------------------------
// fused MLP: out += gelu(xn2 @ fc1^T + b1) @ fc2^T + b2   (32-token blocks)
// ---------------------------------------------------------------------------
__global__ __launch_bounds__(256) void k_mlp(const half_t* __restrict__ xn2,
                                             const half_t* __restrict__ w1,
                                             const float* __restrict__ b1,
                                             const half_t* __restrict__ w2,
                                             const float* __restrict__ b2p,
                                             float* __restrict__ out) {
  __shared__ alignas(32) half_t h1[32][736];
  const int row0 = blockIdx.x * 32;
  const int wv = threadIdx.x >> 5;
  for (int i = threadIdx.x; i < 32 * 16; i += 256) h1[i >> 4][720 + (i & 15)] = (half_t)0;
  // fc1 + GELU: 2 x 45 tiles
  for (int tile = wv; tile < 90; tile += 8) {
    const int rt = tile / 45, ct = tile % 45;
    v8f acc = {};
#pragma unroll
    for (int ks = 0; ks < 6; ++ks) {
      v16h a  = frag_A(xn2, LD, row0 + rt * 16, ks * 32);
      v16h bf = frag_BT(w1, LD, ct * 16, ks * 32);
      acc = wmma_f16(a, bf, acc);
    }
    const int lane = threadIdx.x & 31;
    const int n  = ct * 16 + (lane & 15);
    const int mb = rt * 16 + ((lane >> 4) << 3);
    const float bias = b1[n];
#pragma unroll
    for (int v = 0; v < 8; ++v) {
      float xv = acc[v] + bias;
      h1[mb + v][n] = (half_t)(0.5f * xv * (1.f + erff(xv * 0.70710678118654752f)));
    }
  }
  __syncthreads();
  // fc2 + residual: 2 x 12 tiles, K=736 (padded)
  for (int tile = wv; tile < 24; tile += 8) {
    const int rt = tile / 12, ct = tile % 12;
    v8f acc = {};
    for (int ks = 0; ks < 23; ++ks) {
      v16h a  = frag_A(&h1[0][0], 736, rt * 16, ks * 32);
      v16h bf = frag_BT(w2, 736, ct * 16, ks * 32);
      acc = wmma_f16(a, bf, acc);
    }
    const int lane = threadIdx.x & 31;
    const int oc = ct * 16 + (lane & 15);
    const int mb = row0 + rt * 16 + ((lane >> 4) << 3);
    const float bias = b2p[oc];
    if (oc < DIMC) {
#pragma unroll
      for (int v = 0; v < 8; ++v)
        out[(size_t)(mb + v) * DIMC + oc] += acc[v] + bias;
    }
  }
}

// ---------------------------------------------------------------------------
extern "C" void kernel_launch(void* const* d_in, const int* in_sizes, int n_in,
                              void* d_out, int out_size, void* d_ws, size_t ws_size,
                              hipStream_t stream) {
  (void)in_sizes; (void)n_in; (void)out_size; (void)ws_size;
  const float* x        = (const float*)d_in[0];
  const float* n1g      = (const float*)d_in[1];
  const float* n1b      = (const float*)d_in[2];
  const float* qkv_w    = (const float*)d_in[3];
  const float* q_bias   = (const float*)d_in[4];
  const float* v_bias   = (const float*)d_in[5];
  const float* logit_s  = (const float*)d_in[6];
  const float* cpb_w1   = (const float*)d_in[7];
  const float* cpb_b1   = (const float*)d_in[8];
  const float* cpb_w2   = (const float*)d_in[9];
  const float* proj_w   = (const float*)d_in[10];
  const float* proj_b   = (const float*)d_in[11];
  const float* n2g      = (const float*)d_in[12];
  const float* n2b      = (const float*)d_in[13];
  const float* fc1_w    = (const float*)d_in[14];
  const float* fc1_b    = (const float*)d_in[15];
  const float* fc2_w    = (const float*)d_in[16];
  const float* fc2_b    = (const float*)d_in[17];
  const float* cab_c1_w = (const float*)d_in[18];
  const float* cab_c1_b = (const float*)d_in[19];
  const float* cab_c2_w = (const float*)d_in[20];
  const float* cab_c2_b = (const float*)d_in[21];
  const float* ca_w1    = (const float*)d_in[22];
  const float* ca_b1    = (const float*)d_in[23];
  const float* ca_w2    = (const float*)d_in[24];
  const float* ca_b2    = (const float*)d_in[25];
  float* out = (float*)d_out;

  char* ws = (char*)d_ws;
  size_t off = 0;
  auto alloc = [&](size_t bytes) -> void* {
    off = (off + 255) & ~(size_t)255;
    void* p = ws + off; off += bytes; return p;
  };
  half_t* xn      = (half_t*)alloc((size_t)NTOK * LD * 2);   // reused as xn2
  half_t* attn_o  = (half_t*)alloc((size_t)NTOK * LD * 2);
  half_t* c1      = (half_t*)alloc((size_t)NTOK * 64 * 2);
  half_t* cfeat   = (half_t*)alloc((size_t)NTOK * LD * 2);
  half_t* hqkv    = (half_t*)alloc((size_t)3 * NH * 32 * LD * 2);
  half_t* hproj   = (half_t*)alloc((size_t)LD * LD * 2);
  half_t* hfc1    = (half_t*)alloc((size_t)720 * LD * 2);
  half_t* hfc2    = (half_t*)alloc((size_t)LD * 736 * 2);
  half_t* hw1     = (half_t*)alloc((size_t)9 * 64 * LD * 2);
  half_t* hw2     = (half_t*)alloc((size_t)9 * LD * 64 * 2);
  half_t* zrow    = (half_t*)alloc((size_t)256 * 2);
  float*  bias16  = (float*)alloc((size_t)6 * 64 * 64 * 4);
  float*  ssum    = (float*)alloc((size_t)4 * DIMC * 4);
  float*  aca     = (float*)alloc((size_t)4 * DIMC * 4);
  float*  qkvb    = (float*)alloc((size_t)3 * NH * 32 * 4);
  float*  cb1p    = (float*)alloc((size_t)64 * 4);
  float*  cb2p    = (float*)alloc((size_t)LD * 4);
  float*  pbp     = (float*)alloc((size_t)LD * 4);
  float*  f2bp    = (float*)alloc((size_t)LD * 4);

  // weight / bias prep (padded f16 layouts)
  k_wqkv<<<(3*NH*32*LD + 255) / 256, 256, 0, stream>>>(qkv_w, hqkv);
  k_wpad<<<(LD*LD + 255) / 256, 256, 0, stream>>>(proj_w, hproj, DIMC, DIMC, LD, LD);
  k_wpad<<<(720*LD + 255) / 256, 256, 0, stream>>>(fc1_w, hfc1, 720, DIMC, 720, LD);
  k_wpad<<<(LD*736 + 255) / 256, 256, 0, stream>>>(fc2_w, hfc2, DIMC, 720, LD, 736);
  k_w1r<<<(9*64*LD + 255) / 256, 256, 0, stream>>>(cab_c1_w, hw1);
  k_w2r<<<(9*LD*64 + 255) / 256, 256, 0, stream>>>(cab_c2_w, hw2);
  k_biases<<<3, 256, 0, stream>>>(q_bias, v_bias, cab_c1_b, cab_c2_b, proj_b, fc2_b,
                                  qkvb, cb1p, cb2p, pbp, f2bp);
  k_zero_h<<<1, 256, 0, stream>>>(zrow, 256);
  k_padcols<<<(NTOK*12 + 255) / 256, 256, 0, stream>>>(attn_o);
  k_cpb<<<1, 256, 0, stream>>>(cpb_w1, cpb_b1, cpb_w2, bias16);

  // LN1
  k_ln<<<NTOK/8, 256, 0, stream>>>(x, n1g, n1b, xn);
  // attention (fused qkv/attn/av), one block per (window, head)
  k_attn<<<4096 * NH, 256, 0, stream>>>(xn, hqkv, qkvb, logit_s, bias16, attn_o);
  // CAB branch
  k_conv1<<<NTOK/64, 256, 0, stream>>>(xn, hw1, zrow, cb1p, c1);
  k_conv2<<<NTOK/64, 256, 0, stream>>>(c1, hw2, zrow, cb2p, cfeat);
  k_cmean<<<4 * DIMC, 256, 0, stream>>>(cfeat, ssum);
  k_ca<<<1, 256, 0, stream>>>(ssum, ca_w1, ca_b1, ca_w2, ca_b2, aca);
  // proj + residuals -> d_out (x2)
  k_proj<<<NTOK/64, 256, 0, stream>>>(attn_o, hproj, pbp, x, cfeat, aca, out);
  // LN2 (reuse xn buffer) + fused MLP with in-place residual
  k_ln<<<NTOK/8, 256, 0, stream>>>(out, n2g, n2b, xn);
  k_mlp<<<NTOK/32, 256, 0, stream>>>(xn, hfc1, fc1_b, hfc2, f2bp, out);
}
